// CSPNetFull_49641232007337
// MI455X (gfx1250) — matrix operand order repaired and checked by reading, not compiled, not run on hardware
//
#include <hip/hip_runtime.h>
#include <math.h>

// ---------------------------------------------------------------------------
// CSPNet full forward for MI455X (gfx1250): bf16 WMMA GEMM core with
// Tensor-Data-Mover (TDM) double-buffered tile staging.
// ---------------------------------------------------------------------------

typedef __attribute__((ext_vector_type(16))) __bf16 v16bf;
typedef __attribute__((ext_vector_type(8)))  float  v8f;
typedef unsigned int v4u __attribute__((ext_vector_type(4)));
typedef int          v8i __attribute__((ext_vector_type(8)));
typedef int          v4i __attribute__((ext_vector_type(4)));

#define NB     64        // graphs
#define NPGc   20        // nodes per graph
#define HD     512       // hidden
#define TDd    256       // time dim
#define NFc    128       // fourier freqs
#define NLc    6         // layers
#define Ac     100       // atom types
#define NN     1280      // total nodes
#define NE     24320     // total edges
#define EPGc   380       // edges per graph
#define EINc   2313

__device__ __forceinline__ float sigmoidf_(float x) { return 1.f / (1.f + __expf(-x)); }
__device__ __forceinline__ float siluf_(float x)    { return x / (1.f + __expf(-x)); }

// ---------------------------------------------------------------------------
// TDM: issue a 2D tile load (rows x 16 dwords) from global to LDS.
// Descriptor layout per CDNA5 ISA (D# groups 0/1).
// ---------------------------------------------------------------------------
#if __has_builtin(__builtin_amdgcn_tensor_load_to_lds)
#define USE_TDM 1
#else
#define USE_TDM 0
#endif

__device__ __forceinline__ void tdm_load_tile(unsigned lds_addr, const __bf16* gptr,
                                              int tile_rows, int stride_dw) {
#if USE_TDM
    unsigned long long ga = (unsigned long long)(uintptr_t)gptr;
    const int tile_dw = 16;                    // 32 bf16 = 16 dwords per row
    v4u g0;
    g0[0] = 1u;                                // count=1, user mode
    g0[1] = lds_addr;                          // LDS byte address
    g0[2] = (unsigned)(ga & 0xFFFFFFFFu);      // global_addr[31:0]
    g0[3] = (unsigned)((ga >> 32) & 0x1FFFFFFu) | (2u << 30);  // addr[56:32] | type=2
    v8i g1;
    g1[0] = (int)(2u << 16);                   // workgroup_mask=0, data_size=4B
    g1[1] = (int)((unsigned)tile_dw << 16);    // tensor_dim0 lo16 (== tile_dw)
    g1[2] = (int)((unsigned)(tile_rows & 0xFFFF) << 16);   // dim0 hi | tensor_dim1 lo16
    g1[3] = (int)(((unsigned)tile_rows >> 16) | ((unsigned)tile_dw << 16)); // dim1 hi | tile_dim0
    g1[4] = tile_rows;                         // tile_dim1 | tile_dim2=0
    g1[5] = stride_dw;                         // tensor_dim0_stride (dwords), low 32
    g1[6] = 0;
    g1[7] = 0;
    v4i z4 = {0, 0, 0, 0};
#if __clang_major__ >= 23
    v8i z8 = {0, 0, 0, 0, 0, 0, 0, 0};
    __builtin_amdgcn_tensor_load_to_lds(g0, g1, z4, z4, z8, 0);
#else
    __builtin_amdgcn_tensor_load_to_lds(g0, g1, z4, z4, 0);
#endif
#else
    (void)lds_addr; (void)gptr; (void)tile_rows; (void)stride_dw;
#endif
}

// ---------------------------------------------------------------------------
// Small prep kernels
// ---------------------------------------------------------------------------

__global__ void k_lattice(const float* __restrict__ lp, float* __restrict__ ltl) {
    int b = threadIdx.x;
    if (b >= NB) return;
    const float LM[3] = {1.575442910194397f, 1.7017393112182617f, 1.9781638383865356f};
    const float LS[3] = {0.24437622725963593f, 0.26526379585266113f, 0.3535512685775757f};
    float len[3], c[3], s[3];
#pragma unroll
    for (int i = 0; i < 3; i++) len[i] = __expf(lp[b * 6 + i] * LS[i] + LM[i]);
#pragma unroll
    for (int i = 0; i < 3; i++) {
        float adeg = 59.9f + 60.2f * sigmoidf_(lp[b * 6 + 3 + i]);
        float ar = adeg * 0.017453292519943295f;
        c[i] = cosf(ar); s[i] = sinf(ar);
    }
    float val = (c[0] * c[1] - c[2]) / (s[0] * s[1]);
    val = fminf(1.f, fmaxf(-1.f, val));
    float gs = acosf(val);
    float L[3][3];
    L[0][0] = len[0] * s[1];             L[0][1] = 0.f;                      L[0][2] = len[0] * c[1];
    L[1][0] = -len[1] * s[0] * cosf(gs); L[1][1] = len[1] * s[0] * sinf(gs); L[1][2] = len[1] * c[0];
    L[2][0] = 0.f;                       L[2][1] = 0.f;                      L[2][2] = len[2];
#pragma unroll
    for (int i = 0; i < 3; i++)
#pragma unroll
        for (int j = 0; j < 3; j++)
            ltl[b * 9 + i * 3 + j] = L[i][0] * L[j][0] + L[i][1] * L[j][1] + L[i][2] * L[j][2];
}

__global__ __launch_bounds__(256) void k_embed(const float* __restrict__ Wnode,
                                               const float* __restrict__ Wtime,
                                               const float* __restrict__ Wlat,
                                               const float* __restrict__ tval,
                                               const int* __restrict__ atype,
                                               const int* __restrict__ n2g,
                                               float* __restrict__ h) {
    int n = blockIdx.x;
    int at = atype[n] - 1; at = max(0, min(Ac - 1, at));
    float tv = tval[n2g[n]];
    const float* wrow = Wnode + (size_t)at * HD;
#pragma unroll
    for (int jj = 0; jj < 2; jj++) {
        int j = threadIdx.x + jj * 256;
        float acc = 0.f;
        for (int i = 0; i < HD; i++)  acc += wrow[i] * Wlat[(size_t)i * HD + j];
        for (int i = 0; i < TDd; i++) acc += tv * Wtime[i] * Wlat[(size_t)(HD + i) * HD + j];
        h[(size_t)n * HD + j] = acc;
    }
}

__global__ __launch_bounds__(128) void k_edgepre(const float* __restrict__ fc,
                                                 const int* __restrict__ eidx,
                                                 const float* __restrict__ ltl,
                                                 __bf16* __restrict__ femb,
                                                 float* __restrict__ ud) {
    const float TWO_PI = 6.283185307179586f;
    int e = blockIdx.x, tid = threadIdx.x;
    __shared__ float fd[3];
    int src = eidx[e];
    int dst = eidx[NE + e];
    int g = e / EPGc;
    if (tid < 3) {
        float x = fc[(size_t)src * 3 + tid];
        float y = fc[(size_t)dst * 3 + tid];
        float z = TWO_PI * (y - x);
        fd[tid] = atan2f(sinf(z), cosf(z)) / TWO_PI;
    }
    __syncthreads();
    if (tid == 0) {
        float d[3], nrm = 0.f;
#pragma unroll
        for (int i = 0; i < 3; i++) {
            d[i] = ltl[g * 9 + i * 3 + 0] * fd[0] + ltl[g * 9 + i * 3 + 1] * fd[1] + ltl[g * 9 + i * 3 + 2] * fd[2];
            nrm += d[i] * d[i];
        }
        nrm = sqrtf(nrm) + 1e-12f;
#pragma unroll
        for (int i = 0; i < 3; i++) ud[(size_t)e * 3 + i] = d[i] / nrm;
    }
    for (int i = tid; i < 3 * NFc; i += 128) {
        int c = i >> 7, f = i & (NFc - 1);
        float v = TWO_PI * (float)f * fd[c];
        femb[(size_t)e * 768 + i]       = (__bf16)sinf(v);
        femb[(size_t)e * 768 + 384 + i] = (__bf16)cosf(v);
    }
}

__global__ __launch_bounds__(256) void k_convt(const float* __restrict__ src,
                                               __bf16* __restrict__ dst, int K, int Nout) {
    size_t id = (size_t)blockIdx.x * 256 + threadIdx.x;
    if (id >= (size_t)K * Nout) return;
    int j = (int)(id / K);
    int k = (int)(id % K);
    dst[id] = (__bf16)src[(size_t)k * Nout + j];
}

__global__ __launch_bounds__(256) void k_ln(const float* __restrict__ h,
                                            const float* __restrict__ gamma,
                                            const float* __restrict__ beta,
                                            __bf16* __restrict__ ob, float* __restrict__ of) {
    __shared__ float s1[256], s2[256];
    int n = blockIdx.x, tid = threadIdx.x;
    float x0 = h[(size_t)n * HD + tid];
    float x1 = h[(size_t)n * HD + tid + 256];
    s1[tid] = x0 + x1;
    s2[tid] = x0 * x0 + x1 * x1;
    __syncthreads();
    for (int off = 128; off > 0; off >>= 1) {
        if (tid < off) { s1[tid] += s1[tid + off]; s2[tid] += s2[tid + off]; }
        __syncthreads();
    }
    float mu = s1[0] * (1.f / HD);
    float var = s2[0] * (1.f / HD) - mu * mu;
    float r = rsqrtf(var + 1e-5f);
    float y0 = (x0 - mu) * r * gamma[tid] + beta[tid];
    float y1 = (x1 - mu) * r * gamma[tid + 256] + beta[tid + 256];
    if (of) { of[(size_t)n * HD + tid] = y0; of[(size_t)n * HD + tid + 256] = y1; }
    else    { ob[(size_t)n * HD + tid] = (__bf16)y0; ob[(size_t)n * HD + tid + 256] = (__bf16)y1; }
}

__global__ __launch_bounds__(256) void k_biasg(const float* __restrict__ We1_l,
                                               const float* __restrict__ be1_l,
                                               const float* __restrict__ Wnum_l,
                                               const int* __restrict__ natoms,
                                               const float* __restrict__ lp,
                                               float* __restrict__ biasg) {
    int b = blockIdx.x;
    int na = natoms[b] - 1; na = max(0, min(Ac - 1, na));
    const float* narow = Wnum_l + (size_t)na * HD;
#pragma unroll
    for (int jj = 0; jj < 2; jj++) {
        int j = threadIdx.x + jj * 256;
        float acc = be1_l[j];
        for (int i = 0; i < HD; i++) acc += narow[i] * We1_l[(size_t)(1801 + i) * HD + j];
#pragma unroll
        for (int i = 0; i < 6; i++)  acc += lp[b * 6 + i] * We1_l[(size_t)(1027 + i) * HD + j];
        biasg[(size_t)b * HD + j] = acc;
    }
}

__global__ __launch_bounds__(256) void k_agg(const __bf16* __restrict__ e2,
                                             __bf16* __restrict__ agg) {
    int n = blockIdx.x;
#pragma unroll
    for (int jj = 0; jj < 2; jj++) {
        int j = threadIdx.x + jj * 256;
        float s = 0.f;
        for (int k = 0; k < NPGc - 1; k++)
            s += (float)e2[((size_t)n * (NPGc - 1) + k) * HD + j];
        agg[(size_t)n * HD + j] = (__bf16)(s * (1.f / 19.f));
    }
}

// ---------------------------------------------------------------------------
// WMMA bf16 GEMM: C(MxN) = A(MxK) @ Bt(NxK)^T.
// Block: 128 threads (4 waves). Tile 128x64; each wave computes 32x64
// (2 A-frags x 4 B-frags = 8 WMMAs per 32-wide K step).
// Tiles staged with TDM (double-buffered) when available.
// ---------------------------------------------------------------------------

struct GemmArgs {
    const __bf16* A0; const __bf16* A1;
    int lda0, lda1, Ksplit;
    const __bf16* Bt; int ldb;
    int M, N, K;
    const float* bias;     // per-column, or null
    int mode;              // 0: f32 store, 1: silu->bf16, 2: silu->residual f32 +=, 3: edge1 epilogue
    float* Cf; __bf16* Cb; float* resid;
    // mode 3 extras:
    const float* g;        // N_nodes x 1024 (src proj | dst proj)
    const int* dstIdx;     // E
    const float* biasg;    // B x 512
    const float* ud;       // E x 3
    const float* Wud;      // We1[l] rows 0..2 (3 x 512, f32)
};

// A fragment: 16x32 bf16 (lanes 0-15: K 0-15; lanes 16-31: K 16-31)
__device__ __forceinline__ v16bf load_frag_A(const __bf16* base, int ld) {
    int lane = threadIdx.x & 31;
    int m = lane & 15;
    int kb = (lane < 16) ? 0 : 16;
    const __bf16* p0 = base + m * ld + kb;
    const __bf16* p1 = base + m * ld + 8 + kb;
    v16bf f;
#pragma unroll
    for (int i = 0; i < 8; i++) f[i] = p0[i];
#pragma unroll
    for (int i = 0; i < 8; i++) f[8 + i] = p1[i];
    return f;
}

// B fragment from N-major tile: lanes 0-15 hold K 0..15 of col n=lane; lanes 16-31 K 16..31
__device__ __forceinline__ v16bf load_frag_B(const __bf16* base, int ld) {
    int lane = threadIdx.x & 31;
    int n = lane & 15;
    int kb = (lane < 16) ? 0 : 16;
    const __bf16* p = base + n * ld + kb;
    v16bf f;
#pragma unroll
    for (int i = 0; i < 16; i++) f[i] = p[i];
    return f;
}

__global__ __launch_bounds__(128) void k_gemm(GemmArgs a) {
    __shared__ __align__(16) __bf16 As[2][128][32];
    __shared__ __align__(16) __bf16 Bs[2][64][32];
    int row0 = blockIdx.x * 128;
    int col0 = blockIdx.y * 64;
    int tid = threadIdx.x;
    int wave = tid >> 5;
    int lane = tid & 31;
    v8f acc[2][4] = {};

    // Prologue: stage tile 0 into buffer 0 (wave 0 drives the TDM).
    if (wave == 0) {
        const __bf16* Ap = a.A0; int lda = a.lda0;
        tdm_load_tile((unsigned)(uintptr_t)&As[0][0][0],
                      Ap + (size_t)row0 * lda, 128, lda / 2);
        tdm_load_tile((unsigned)(uintptr_t)&Bs[0][0][0],
                      a.Bt + (size_t)col0 * a.ldb, 64, a.ldb / 2);
    }

    const int nk = a.K / 32;
    for (int i = 0; i < nk; i++) {
        int buf = i & 1;
        if (wave == 0) {
            if (i + 1 < nk) {
                int k0 = (i + 1) * 32;
                const __bf16* Ap; int lda, kk;
                if (k0 < a.Ksplit) { Ap = a.A0; lda = a.lda0; kk = k0; }
                else               { Ap = a.A1; lda = a.lda1; kk = k0 - a.Ksplit; }
                tdm_load_tile((unsigned)(uintptr_t)&As[buf ^ 1][0][0],
                              Ap + (size_t)row0 * lda + kk, 128, lda / 2);
                tdm_load_tile((unsigned)(uintptr_t)&Bs[buf ^ 1][0][0],
                              a.Bt + (size_t)col0 * a.ldb + k0, 64, a.ldb / 2);
#if USE_TDM
                __builtin_amdgcn_s_wait_tensorcnt(2);  // next 2 outstanding; current tile done
#endif
            } else {
#if USE_TDM
                __builtin_amdgcn_s_wait_tensorcnt(0);
#endif
            }
        }
        __syncthreads();
        v16bf af0 = load_frag_A(&As[buf][wave * 32][0], 32);
        v16bf af1 = load_frag_A(&As[buf][wave * 32 + 16][0], 32);
#pragma unroll
        for (int c = 0; c < 4; c++) {
            v16bf bf = load_frag_B(&Bs[buf][c * 16][0], 32);
            acc[0][c] = __builtin_amdgcn_wmma_f32_16x16x32_bf16(
                false, af0, false, bf, (short)0, acc[0][c], false, false);
            acc[1][c] = __builtin_amdgcn_wmma_f32_16x16x32_bf16(
                false, af1, false, bf, (short)0, acc[1][c], false, false);
        }
        __syncthreads();
    }

    // Epilogue. C layout: VGPR r -> M = r + (lane>=16 ? 8 : 0), N = lane & 15.
    int hi8 = (lane & 16) ? 8 : 0;
    int ncol = lane & 15;
#pragma unroll
    for (int sub = 0; sub < 2; sub++) {
#pragma unroll
        for (int c = 0; c < 4; c++) {
            int gcol = col0 + c * 16 + ncol;
#pragma unroll
            for (int r = 0; r < 8; r++) {
                int grow = row0 + wave * 32 + sub * 16 + hi8 + r;
                float v = acc[sub][c][r];
                if (a.bias) v += a.bias[gcol];
                if (a.mode == 3) {
                    int e = grow;
                    int srcn = e / 19;
                    int grp  = e / EPGc;
                    int dstn = a.dstIdx[e];
                    v += a.g[(size_t)srcn * 1024 + gcol];
                    v += a.g[(size_t)dstn * 1024 + 512 + gcol];
                    v += a.biasg[(size_t)grp * HD + gcol];
                    const float* u = a.ud + (size_t)e * 3;
                    v += u[0] * a.Wud[gcol] + u[1] * a.Wud[HD + gcol] + u[2] * a.Wud[2 * HD + gcol];
                }
                if (a.mode >= 1) v = siluf_(v);
                if (a.mode == 0)      a.Cf[(size_t)grow * a.N + gcol] = v;
                else if (a.mode == 2) a.resid[(size_t)grow * a.N + gcol] += v;
                else                  a.Cb[(size_t)grow * a.N + gcol] = (__bf16)v;
            }
        }
    }
}

// ---------------------------------------------------------------------------
// Output heads
// ---------------------------------------------------------------------------

__global__ __launch_bounds__(128) void k_pos(const float* __restrict__ hf,
                                             const float* __restrict__ Wcoord,
                                             float* __restrict__ out) {
    int id = blockIdx.x * 128 + threadIdx.x;
    if (id >= NN * 3) return;
    int n = id / 3, c = id % 3;
    float s = 0.f;
    for (int j = 0; j < HD; j++) s += hf[(size_t)n * HD + j] * Wcoord[j * 3 + c];
    out[id] = s;
}

__global__ __launch_bounds__(256) void k_cell(const float* __restrict__ hf,
                                              const float* __restrict__ Wlout,
                                              const float* __restrict__ blout,
                                              const int* __restrict__ natoms,
                                              float* __restrict__ out) {
    __shared__ float s[HD];
    int b = blockIdx.x, tid = threadIdx.x;
#pragma unroll
    for (int jj = 0; jj < 2; jj++) {
        int j = tid + jj * 256;
        float ss = 0.f;
        for (int m = 0; m < NPGc; m++) ss += hf[((size_t)b * NPGc + m) * HD + j];
        s[j] = ss;
    }
    __syncthreads();
    if (tid < 6) {
        float inv = 1.f / (float)natoms[b];
        float acc = blout[tid];
        for (int j = 0; j < HD; j++)
            acc += (s[j] * inv) * Wlout[j * 6 + tid] + s[j] * Wlout[(HD + j) * 6 + tid];
        out[NN * 3 + b * 6 + tid] = acc;
    }
}

// ---------------------------------------------------------------------------
// Host orchestration
// ---------------------------------------------------------------------------

extern "C" void kernel_launch(void* const* d_in, const int* in_sizes, int n_in,
                              void* d_out, int out_size, void* d_ws, size_t ws_size,
                              hipStream_t stream) {
    (void)in_sizes; (void)n_in; (void)out_size; (void)ws_size;
    const float* t_in  = (const float*)d_in[0];
    const float* fc    = (const float*)d_in[1];
    const float* lp    = (const float*)d_in[2];
    const int*   atype = (const int*)d_in[3];
    const int*   natom = (const int*)d_in[4];
    const int*   n2g   = (const int*)d_in[5];
    const int*   eidx  = (const int*)d_in[6];
    const float* Wnode = (const float*)d_in[8];
    const float* Wtime = (const float*)d_in[9];
    const float* Wlat  = (const float*)d_in[10];
    const float* lng   = (const float*)d_in[11];
    const float* lnb   = (const float*)d_in[12];
    const float* We1   = (const float*)d_in[13];
    const float* be1   = (const float*)d_in[14];
    const float* We2   = (const float*)d_in[15];
    const float* be2   = (const float*)d_in[16];
    const float* Wn1   = (const float*)d_in[17];
    const float* bn1   = (const float*)d_in[18];
    const float* Wn2   = (const float*)d_in[19];
    const float* bn2   = (const float*)d_in[20];
    const float* Wnum  = (const float*)d_in[21];
    const float* flng  = (const float*)d_in[22];
    const float* flnb  = (const float*)d_in[23];
    const float* Wcrd  = (const float*)d_in[24];
    const float* Wlout = (const float*)d_in[25];
    const float* blout = (const float*)d_in[26];

    char* base = (char*)d_ws;
    size_t off = 0;
    auto alloc = [&](size_t bytes) -> void* {
        void* p = base + off;
        off += (bytes + 255) & ~(size_t)255;
        return p;
    };
    __bf16* femb  = (__bf16*)alloc((size_t)NE * 768 * 2);
    float*  ud    = (float*) alloc((size_t)NE * 3 * 4);
    float*  ltl   = (float*) alloc((size_t)NB * 9 * 4);
    float*  h     = (float*) alloc((size_t)NN * HD * 4);
    __bf16* hn    = (__bf16*)alloc((size_t)NN * HD * 2);
    float*  g     = (float*) alloc((size_t)NN * 1024 * 4);
    float*  biasg = (float*) alloc((size_t)NB * HD * 4);
    __bf16* t1    = (__bf16*)alloc((size_t)NE * HD * 2);
    __bf16* e2b   = (__bf16*)alloc((size_t)NE * HD * 2);
    __bf16* aggb  = (__bf16*)alloc((size_t)NN * HD * 2);
    __bf16* nm1   = (__bf16*)alloc((size_t)NN * HD * 2);
    float*  hf    = (float*) alloc((size_t)NN * HD * 4);
    __bf16* Wab[NLc], *Wf[NLc], *We2t[NLc], *Wn1t[NLc], *Wn2t[NLc];
    for (int l = 0; l < NLc; l++) {
        Wab[l]  = (__bf16*)alloc((size_t)1024 * 512 * 2);
        Wf[l]   = (__bf16*)alloc((size_t)512 * 768 * 2);
        We2t[l] = (__bf16*)alloc((size_t)512 * 512 * 2);
        Wn1t[l] = (__bf16*)alloc((size_t)512 * 1024 * 2);
        Wn2t[l] = (__bf16*)alloc((size_t)512 * 512 * 2);
    }

    auto convt = [&](const float* src, __bf16* dst, int K, int Nout) {
        int total = K * Nout;
        k_convt<<<(total + 255) / 256, 256, 0, stream>>>(src, dst, K, Nout);
    };

    // Weight conversion (pre-transposed to N-major bf16)
    for (int l = 0; l < NLc; l++) {
        const float* We1_l = We1 + (size_t)l * EINc * HD;
        convt(We1_l + (size_t)3 * HD,      Wab[l],             HD,   HD);  // src-proj Wa
        convt(We1_l + (size_t)515 * HD,    Wab[l] + 512 * 512, HD,   HD);  // dst-proj Wb
        convt(We1_l + (size_t)1033 * HD,   Wf[l],              768,  HD);  // fourier part
        convt(We2 + (size_t)l * HD * HD,   We2t[l],            HD,   HD);
        convt(Wn1 + (size_t)l * 1024 * HD, Wn1t[l],            1024, HD);
        convt(Wn2 + (size_t)l * HD * HD,   Wn2t[l],            HD,   HD);
    }

    // Graph / node / edge precompute
    k_lattice<<<1, 64, 0, stream>>>(lp, ltl);
    k_embed<<<NN, 256, 0, stream>>>(Wnode, Wtime, Wlat, t_in, atype, n2g, h);
    k_edgepre<<<NE, 128, 0, stream>>>(fc, eidx, ltl, femb, ud);

    for (int l = 0; l < NLc; l++) {
        const float* We1_l = We1 + (size_t)l * EINc * HD;

        k_ln<<<NN, 256, 0, stream>>>(h, lng + l * HD, lnb + l * HD, hn, nullptr);

        // g = hn @ [Wa | Wb]  (1280 x 512 x 1024), f32 out
        GemmArgs ga = {};
        ga.A0 = hn; ga.A1 = hn; ga.lda0 = HD; ga.lda1 = HD; ga.Ksplit = 1 << 30;
        ga.Bt = Wab[l]; ga.ldb = HD; ga.M = NN; ga.N = 1024; ga.K = HD;
        ga.mode = 0; ga.Cf = g;
        k_gemm<<<dim3(NN / 128, 1024 / 64), 128, 0, stream>>>(ga);

        k_biasg<<<NB, 256, 0, stream>>>(We1_l, be1 + l * HD, Wnum + (size_t)l * Ac * HD,
                                        natom, lp, biasg);

        // t1 = silu(femb @ Wf + g[src] + g[dst] + ud@Wud + biasg)  (24320 x 768 x 512)
        GemmArgs ge1 = {};
        ge1.A0 = femb; ge1.A1 = femb; ge1.lda0 = 768; ge1.lda1 = 768; ge1.Ksplit = 1 << 30;
        ge1.Bt = Wf[l]; ge1.ldb = 768; ge1.M = NE; ge1.N = HD; ge1.K = 768;
        ge1.mode = 3; ge1.Cb = t1;
        ge1.g = g; ge1.dstIdx = eidx + NE; ge1.biasg = biasg; ge1.ud = ud; ge1.Wud = We1_l;
        k_gemm<<<dim3(NE / 128, HD / 64), 128, 0, stream>>>(ge1);

        // e2 = silu(t1 @ We2 + be2)  (24320 x 512 x 512)
        GemmArgs ge2 = {};
        ge2.A0 = t1; ge2.A1 = t1; ge2.lda0 = HD; ge2.lda1 = HD; ge2.Ksplit = 1 << 30;
        ge2.Bt = We2t[l]; ge2.ldb = HD; ge2.M = NE; ge2.N = HD; ge2.K = HD;
        ge2.mode = 1; ge2.bias = be2 + l * HD; ge2.Cb = e2b;
        k_gemm<<<dim3(NE / 128, HD / 64), 128, 0, stream>>>(ge2);

        k_agg<<<NN, 256, 0, stream>>>(e2b, aggb);

        // nm1 = silu([hn | agg] @ Wn1 + bn1)  (1280 x 1024 x 512)
        GemmArgs gn1 = {};
        gn1.A0 = hn; gn1.A1 = aggb; gn1.lda0 = HD; gn1.lda1 = HD; gn1.Ksplit = HD;
        gn1.Bt = Wn1t[l]; gn1.ldb = 1024; gn1.M = NN; gn1.N = HD; gn1.K = 1024;
        gn1.mode = 1; gn1.bias = bn1 + l * HD; gn1.Cb = nm1;
        k_gemm<<<dim3(NN / 128, HD / 64), 128, 0, stream>>>(gn1);

        // h += silu(nm1 @ Wn2 + bn2)  (1280 x 512 x 512)
        GemmArgs gn2 = {};
        gn2.A0 = nm1; gn2.A1 = nm1; gn2.lda0 = HD; gn2.lda1 = HD; gn2.Ksplit = 1 << 30;
        gn2.Bt = Wn2t[l]; gn2.ldb = HD; gn2.M = NN; gn2.N = HD; gn2.K = HD;
        gn2.mode = 2; gn2.bias = bn2 + l * HD; gn2.resid = h;
        k_gemm<<<dim3(NN / 128, HD / 64), 128, 0, stream>>>(gn2);
    }

    // Final LN + heads
    k_ln<<<NN, 256, 0, stream>>>(h, flng, flnb, nullptr, hf);
    k_pos<<<(NN * 3 + 127) / 128, 128, 0, stream>>>(hf, Wcrd, (float*)d_out);
    k_cell<<<NB, 256, 0, stream>>>(hf, Wlout, blout, natom, (float*)d_out);
}
